// GPT_49280454754587
// MI455X (gfx1250) — matrix operand order repaired and checked by reading, not compiled, not run on hardware
//
#include <hip/hip_runtime.h>
#include <hip/hip_bf16.h>

typedef __attribute__((ext_vector_type(16))) __bf16 v16bf;
typedef __attribute__((ext_vector_type(8)))  float  v8f;

namespace {

constexpr int kLayers = 12;
constexpr int kHeads  = 12;
constexpr int kD      = 768;
constexpr int kDH     = 64;
constexpr int kV      = 50257;
constexpr int kT      = 1024;
constexpr int kB      = 4;
constexpr int kM      = kB * kT;   // 4096 token rows
constexpr int kD3     = 3 * kD;    // 2304
constexpr int kD4     = 4 * kD;    // 3072

__device__ __forceinline__ __bf16 f2bf(float f) {
  unsigned u = __float_as_uint(f);
  u += 0x7FFFu + ((u >> 16) & 1u);              // round-to-nearest-even
  unsigned short s = (unsigned short)(u >> 16);
  __bf16 r;
  __builtin_memcpy(&r, &s, sizeof(r));
  return r;
}

union Frag16 {            // 16 bf16 per lane = 8 VGPRs; two 16B chunks
  v16bf v;
  uint4 q[2];
};

__device__ __forceinline__ v8f vzero8() {
  v8f z;
  #pragma unroll
  for (int i = 0; i < 8; ++i) z[i] = 0.0f;
  return z;
}

__device__ __forceinline__ v8f wmma_bf16(v16bf a, v16bf b, v8f c) {
  return __builtin_amdgcn_wmma_f32_16x16x32_bf16(false, a, false, b, (short)0, c,
                                                 false, false);
}

// reduce across the 16 lanes of each half-wave (xor masks < 16 keep bit 4)
__device__ __forceinline__ float redmax16(float v) {
  #pragma unroll
  for (int m = 1; m < 16; m <<= 1) v = fmaxf(v, __shfl_xor(v, m, 32));
  return v;
}
__device__ __forceinline__ float redsum16(float v) {
  #pragma unroll
  for (int m = 1; m < 16; m <<= 1) v += __shfl_xor(v, m, 32);
  return v;
}

// ---------------------------------------------------------------- embedding
__global__ void embed_kernel(const int* __restrict__ tokens,
                             const float* __restrict__ wte,
                             const float* __restrict__ wpe,
                             float* __restrict__ x) {
  int idx = blockIdx.x * blockDim.x + threadIdx.x;
  if (idx >= kM * kD) return;
  int row = idx / kD;
  int c   = idx - row * kD;
  int t   = row & (kT - 1);
  int tok = tokens[row];
  x[idx] = wte[(size_t)tok * kD + c] + wpe[(size_t)t * kD + c];
}

// ---------------------------------------------------------------- layernorm
// one wave32 per row of D=768; writes bf16 (GEMM A operand)
__global__ void ln_kernel(const float* __restrict__ x,
                          const float* __restrict__ g,
                          const float* __restrict__ b,
                          __bf16* __restrict__ out, int rows) {
  int wave = (int)((blockIdx.x * blockDim.x + threadIdx.x) >> 5);
  int lane = threadIdx.x & 31;
  if (wave >= rows) return;
  const float* xr = x + (size_t)wave * kD;
  float s = 0.f, s2 = 0.f;
  for (int c = lane; c < kD; c += 32) {
    float v = xr[c];
    s += v;
    s2 += v * v;
  }
  #pragma unroll
  for (int m = 1; m < 32; m <<= 1) {
    s  += __shfl_xor(s,  m, 32);
    s2 += __shfl_xor(s2, m, 32);
  }
  float mean = s * (1.0f / kD);
  float var  = s2 * (1.0f / kD) - mean * mean;
  float inv  = rsqrtf(var + 1e-5f);
  __bf16* orow = out + (size_t)wave * kD;
  for (int c = lane; c < kD; c += 32)
    orow[c] = f2bf((xr[c] - mean) * inv * g[c] + b[c]);
}

// ---------------------------------------------------------------- B pre-pack
// fp32 weights [K,N] row-major -> bf16 in B-fragment-native order:
// P[(k/32)*N*32 + n*32 + (k%32)], so a lane's 16 K-values are contiguous.
__global__ void pack_b_kernel(const float* __restrict__ Bw,
                              __bf16* __restrict__ P, int K, int N) {
  size_t idx = (size_t)blockIdx.x * blockDim.x + threadIdx.x;
  size_t total = (size_t)K * N;
  if (idx >= total) return;
  int k = (int)(idx / N);
  int n = (int)(idx - (size_t)k * N);
  P[((size_t)(k >> 5) * N + n) * 32 + (k & 31)] = f2bf(Bw[idx]);
}

// ---------------------------------------------------------------- GEMM
// C[M,N] = A[M,K](bf16) @ Bp (pre-packed bf16) + bias
// 256 threads = 8 waves tiled 2x4; each wave owns a 32x32 C tile
// (2x2 WMMA frags). Block tile: 64 x 128. All operand loads are b128.
// MODE 0: store bf16(val+bias)            (qkv)
// MODE 1: out(f32)[m,n] += val+bias       (residual-accumulating projections)
// MODE 2: store bf16(gelu(val+bias))      (mlp fc)
// MODE 3: store f32(val)                  (lm head)
template <int MODE>
__global__ void __launch_bounds__(256)
gemm_wmma(const __bf16* __restrict__ A, const __bf16* __restrict__ Bp,
          const float* __restrict__ bias, void* __restrict__ outp,
          int M, int N, int K) {
  const int lane  = threadIdx.x & 31;
  const int wid   = threadIdx.x >> 5;         // 0..7
  const int lmod  = lane & 15;
  const int lhalf = lane >> 4;                // 0/1
  const int rb = blockIdx.y * 64  + (wid & 1) * 32;
  const int cb = blockIdx.x * 128 + (wid >> 1) * 32;

  v8f acc[2][2];
  #pragma unroll
  for (int i = 0; i < 2; ++i)
    #pragma unroll
    for (int j = 0; j < 2; ++j) acc[i][j] = vzero8();

  for (int kk = 0; kk < K; kk += 32) {
    // A fragments: 16x32 bf16, lane m=l%16, K = lhalf*16 + e (contiguous)
    Frag16 a[2];
    #pragma unroll
    for (int i = 0; i < 2; ++i) {
      const __bf16* ap = A + (size_t)(rb + i * 16 + lmod) * K + kk + lhalf * 16;
      a[i].q[0] = *(const uint4*)ap;
      a[i].q[1] = *(const uint4*)(ap + 8);
    }
    // B fragments: 32x16, lane n=l%16, K = lhalf*16 + e (contiguous packed)
    Frag16 b[2];
    #pragma unroll
    for (int j = 0; j < 2; ++j) {
      int n = cb + j * 16 + lmod;
      if (n < N) {
        const __bf16* bp =
            Bp + ((size_t)(kk >> 5) * N + n) * 32 + lhalf * 16;
        b[j].q[0] = *(const uint4*)bp;
        b[j].q[1] = *(const uint4*)(bp + 8);
      } else {
        b[j].q[0] = make_uint4(0u, 0u, 0u, 0u);
        b[j].q[1] = make_uint4(0u, 0u, 0u, 0u);
      }
    }
    #pragma unroll
    for (int i = 0; i < 2; ++i)
      #pragma unroll
      for (int j = 0; j < 2; ++j)
        acc[i][j] = wmma_bf16(a[i].v, b[j].v, acc[i][j]);
  }

  // epilogue: C layout lane n=l%16, row m = r + lhalf*8
  #pragma unroll
  for (int j = 0; j < 2; ++j) {
    int n = cb + j * 16 + lmod;
    if (n >= N) continue;
    float bv = (MODE == 3) ? 0.0f : bias[n];
    #pragma unroll
    for (int i = 0; i < 2; ++i) {
      #pragma unroll
      for (int r = 0; r < 8; ++r) {
        int m = rb + i * 16 + lhalf * 8 + r;
        float val = acc[i][j][r] + bv;
        size_t oi = (size_t)m * N + n;
        if (MODE == 0) {
          ((__bf16*)outp)[oi] = f2bf(val);
        } else if (MODE == 1) {
          ((float*)outp)[oi] += val;
        } else if (MODE == 2) {
          float gel = 0.5f * val * (1.0f + erff(val * 0.70710678118654752f));
          ((__bf16*)outp)[oi] = f2bf(gel);
        } else {
          ((float*)outp)[oi] = val;
        }
      }
    }
  }
}

// ---------------------------------------------------------------- attention
// Flash-style: one wave32 per (b, h, 16-query tile); streams 32-key chunks.
// S = Q K^T via 4 WMMA; online softmax via half-wave shuffles; P staged
// through LDS (C-layout -> A-layout transpose); O += P V via 4 WMMA.
// All strided gathers use ONE per-lane base pointer + compile-time
// immediate offsets (e*4608B fits the signed 24-bit VGLOBAL offset), so no
// per-element 64-bit address math in the hot loop.
__global__ void __launch_bounds__(32, 1)
attn_kernel(const __bf16* __restrict__ qkv, __bf16* __restrict__ o) {
  __shared__ __align__(16) __bf16 pbuf[16 * 32];   // one wave per block
  const int lane  = threadIdx.x & 31;
  const int lmod  = lane & 15;
  const int lhalf = lane >> 4;
  const int qt = blockIdx.x;        // 0..63
  const int hh = blockIdx.y;        // 0..11
  const int bb = blockIdx.z;        // 0..3
  const int qbase = qt * 16;
  const int rstride = kD3;          // 2304 bf16 per token row of qkv
  const __bf16* base = qkv + (size_t)(bb * kT) * rstride + hh * kDH;

  // Q fragments (A layout, DH=64 -> two 16x32 frags), contiguous 16B loads
  Frag16 aq[2];
  {
    const __bf16* qp = base + (size_t)(qbase + lmod) * rstride;   // Q at +0
    #pragma unroll
    for (int i = 0; i < 2; ++i) {
      const __bf16* pp = qp + i * 32 + lhalf * 16;
      aq[i].q[0] = *(const uint4*)pp;
      aq[i].q[1] = *(const uint4*)(pp + 8);
    }
  }

  v8f oacc[4];
  #pragma unroll
  for (int j = 0; j < 4; ++j) oacc[j] = vzero8();
  float rm[8], rs[8];
  #pragma unroll
  for (int r = 0; r < 8; ++r) { rm[r] = -1e30f; rs[r] = 0.0f; }

  // per-lane base pointers, advanced by 32 rows per chunk iteration
  const __bf16* kptr = base + (size_t)lmod * rstride + kD;            // K
  const __bf16* vptr = base + (size_t)(lhalf * 16) * rstride + 2 * kD; // V

  for (int kb = 0; kb < qbase + 16; kb += 32) {
    // ---- S = Q K^T for 32 keys (two 16x16 tiles)
    v8f s[2];
    #pragma unroll
    for (int jt = 0; jt < 2; ++jt) {
      s[jt] = vzero8();
      // B operand: n = key (l%16), k-dim = d = i*32 + lhalf*16 + e (contig!)
      const __bf16* kp = kptr + (size_t)(jt * 16) * rstride;
      Frag16 bk[2];
      #pragma unroll
      for (int i = 0; i < 2; ++i) {
        const __bf16* pp = kp + i * 32 + lhalf * 16;
        bk[i].q[0] = *(const uint4*)pp;
        bk[i].q[1] = *(const uint4*)(pp + 8);
      }
      s[jt] = wmma_bf16(aq[0].v, bk[0].v, s[jt]);
      s[jt] = wmma_bf16(aq[1].v, bk[1].v, s[jt]);
    }

    // ---- online softmax update (rows m = lhalf*8 + r, cols = keys)
    float pj[2][8], tmax[8];
    #pragma unroll
    for (int r = 0; r < 8; ++r) {
      int qrow = qbase + lhalf * 8 + r;
      float v0 = s[0][r] * 0.125f;               // 1/sqrt(64)
      float v1 = s[1][r] * 0.125f;
      if (kb + lmod > qrow)       v0 = -1e30f;   // causal mask
      if (kb + 16 + lmod > qrow)  v1 = -1e30f;
      pj[0][r] = v0; pj[1][r] = v1;
      tmax[r] = redmax16(fmaxf(v0, v1));
    }
    #pragma unroll
    for (int r = 0; r < 8; ++r) {
      float mn = fmaxf(rm[r], tmax[r]);
      float sc = __expf(rm[r] - mn);
      float p0 = __expf(pj[0][r] - mn);
      float p1 = __expf(pj[1][r] - mn);
      pj[0][r] = p0; pj[1][r] = p1;
      rs[r] = rs[r] * sc + redsum16(p0 + p1);
      rm[r] = mn;
      #pragma unroll
      for (int j = 0; j < 4; ++j) oacc[j][r] *= sc;
    }

    // ---- transpose P (C layout) -> A layout via LDS
    #pragma unroll
    for (int r = 0; r < 8; ++r) {
      int m = lhalf * 8 + r;
      pbuf[m * 32 + lmod]      = f2bf(pj[0][r]);
      pbuf[m * 32 + 16 + lmod] = f2bf(pj[1][r]);
    }
    asm volatile("s_wait_dscnt 0" ::: "memory");   // in-wave DS ordering fence
    Frag16 ap;
    {
      const __bf16* pp = &pbuf[lmod * 32 + lhalf * 16];
      ap.q[0] = *(const uint4*)pp;
      ap.q[1] = *(const uint4*)(pp + 8);
    }

    // ---- O += P V : one base pointer, constant row offsets (e * 4608 B)
    #pragma unroll
    for (int j = 0; j < 4; ++j) {
      v16bf bv;
      #pragma unroll
      for (int e = 0; e < 16; ++e)
        bv[e] = vptr[(size_t)e * rstride + j * 16 + lmod];
      oacc[j] = wmma_bf16(ap.v, bv, oacc[j]);
    }

    kptr += (size_t)32 * rstride;
    vptr += (size_t)32 * rstride;
  }

  // ---- normalize and write O (bf16, [B*T, 768] at head offset)
  #pragma unroll
  for (int r = 0; r < 8; ++r) {
    float inv = 1.0f / rs[r];
    int q = qbase + lhalf * 8 + r;
    __bf16* orow = o + (size_t)(bb * kT + q) * kD + hh * kDH;
    #pragma unroll
    for (int j = 0; j < 4; ++j) orow[j * 16 + lmod] = f2bf(oacc[j][r] * inv);
  }
}

}  // namespace

extern "C" void kernel_launch(void* const* d_in, const int* in_sizes, int n_in,
                              void* d_out, int out_size, void* d_ws,
                              size_t ws_size, hipStream_t stream) {
  (void)in_sizes; (void)n_in; (void)out_size; (void)ws_size;

  const int*   tokens  = (const int*)  d_in[0];
  const float* wte     = (const float*)d_in[1];
  const float* wpe     = (const float*)d_in[2];
  const float* ln1_g   = (const float*)d_in[3];
  const float* ln1_b   = (const float*)d_in[4];
  const float* attn_w  = (const float*)d_in[5];
  const float* attn_b  = (const float*)d_in[6];
  const float* aproj_w = (const float*)d_in[7];
  const float* aproj_b = (const float*)d_in[8];
  const float* ln2_g   = (const float*)d_in[9];
  const float* ln2_b   = (const float*)d_in[10];
  const float* fc_w    = (const float*)d_in[11];
  const float* fc_b    = (const float*)d_in[12];
  const float* mproj_w = (const float*)d_in[13];
  const float* mproj_b = (const float*)d_in[14];
  const float* lnf_g   = (const float*)d_in[15];
  const float* lnf_b   = (const float*)d_in[16];
  const float* head_w  = (const float*)d_in[17];

  // workspace carve-up (all chunks naturally 256B-aligned)
  char* ws = (char*)d_ws;
  float*  x    = (float*)ws;  ws += (size_t)kM * kD  * sizeof(float);
  __bf16* h    = (__bf16*)ws; ws += (size_t)kM * kD  * sizeof(__bf16);
  __bf16* qkvb = (__bf16*)ws; ws += (size_t)kM * kD3 * sizeof(__bf16);
  __bf16* oatt = (__bf16*)ws; ws += (size_t)kM * kD  * sizeof(__bf16);
  __bf16* gact = (__bf16*)ws; ws += (size_t)kM * kD4 * sizeof(__bf16);
  __bf16* bpk  = (__bf16*)ws;                  // reusable packed-B buffer:
  // sized for the largest weight matrix (head: 768 x 50257 ~ 77 MB bf16)

  const dim3 gThr(256);
  const dim3 lnGrid(kM / 8);            // 8 waves (rows) per 256-thread block

  auto packGrid = [](long long kn) { return dim3((unsigned)((kn + 255) / 256)); };

  embed_kernel<<<(kM * kD + 255) / 256, gThr, 0, stream>>>(tokens, wte, wpe, x);

  for (int l = 0; l < kLayers; ++l) {
    // attn: ln1 -> qkv -> flash attn -> proj(+residual)
    ln_kernel<<<lnGrid, gThr, 0, stream>>>(x, ln1_g + (size_t)l * kD,
                                           ln1_b + (size_t)l * kD, h, kM);
    pack_b_kernel<<<packGrid((long long)kD * kD3), gThr, 0, stream>>>(
        attn_w + (size_t)l * kD * kD3, bpk, kD, kD3);
    gemm_wmma<0><<<dim3((kD3 + 127) / 128, kM / 64), gThr, 0, stream>>>(
        h, bpk, attn_b + (size_t)l * kD3, qkvb, kM, kD3, kD);
    attn_kernel<<<dim3(kT / 16, kHeads, kB), dim3(32), 0, stream>>>(qkvb, oatt);
    pack_b_kernel<<<packGrid((long long)kD * kD), gThr, 0, stream>>>(
        aproj_w + (size_t)l * kD * kD, bpk, kD, kD);
    gemm_wmma<1><<<dim3((kD + 127) / 128, kM / 64), gThr, 0, stream>>>(
        oatt, bpk, aproj_b + (size_t)l * kD, x, kM, kD, kD);

    // mlp: ln2 -> fc(+gelu) -> proj(+residual)
    ln_kernel<<<lnGrid, gThr, 0, stream>>>(x, ln2_g + (size_t)l * kD,
                                           ln2_b + (size_t)l * kD, h, kM);
    pack_b_kernel<<<packGrid((long long)kD * kD4), gThr, 0, stream>>>(
        fc_w + (size_t)l * kD * kD4, bpk, kD, kD4);
    gemm_wmma<2><<<dim3((kD4 + 127) / 128, kM / 64), gThr, 0, stream>>>(
        h, bpk, fc_b + (size_t)l * kD4, gact, kM, kD4, kD);
    pack_b_kernel<<<packGrid((long long)kD4 * kD), gThr, 0, stream>>>(
        mproj_w + (size_t)l * kD4 * kD, bpk, kD4, kD);
    gemm_wmma<1><<<dim3((kD + 127) / 128, kM / 64), gThr, 0, stream>>>(
        gact, bpk, mproj_b + (size_t)l * kD, x, kM, kD, kD4);
  }

  // final layernorm + lm head
  ln_kernel<<<lnGrid, gThr, 0, stream>>>(x, lnf_g, lnf_b, h, kM);
  pack_b_kernel<<<packGrid((long long)kD * kV), gThr, 0, stream>>>(
      head_w, bpk, kD, kV);
  gemm_wmma<3><<<dim3((kV + 127) / 128, kM / 64), gThr, 0, stream>>>(
      h, bpk, nullptr, d_out, kM, kV, kD);
}
